// Attention_49125835932028
// MI455X (gfx1250) — matrix operand order repaired and checked by reading, not compiled, not run on hardware
//
#include <hip/hip_runtime.h>

// ---------------------------------------------------------------------------
// MI455X (gfx1250, wave32) fused joint-attention block.
// All GEMM-like math through v_wmma_f32_16x16x32_bf16 (f32 accumulate).
// ---------------------------------------------------------------------------

typedef __bf16 bf16;
typedef __attribute__((ext_vector_type(16))) __bf16 v16bf;
typedef __attribute__((ext_vector_type(2)))  __bf16 v2bf;
typedef __attribute__((ext_vector_type(8)))  float  v8f;

#define D_MODEL 3072
#define DH      128
#define NH      24
#define S_ALL   2560
#define S_IMG   2048
#define S_TXT   512

#define KSTEP 64
// LDS strides (padded against bank conflicts)
#define AS  72   // A tile row stride, bf16 units (64 + 8 pad)
#define BTS 72   // B^T tile row stride, bf16 units (64 k + 8 pad)
#define CS  132  // C tile row stride, f32 units (128 + 4 pad)
#define PS  34   // P staging row stride, f32 units (32 + 2 pad)

#if defined(__has_builtin)
# if __has_builtin(__builtin_amdgcn_cvt_pk_bf16_f32)
#  define HAVE_PK_BF16 1
# endif
#endif
#ifndef HAVE_PK_BF16
# define HAVE_PK_BF16 0
#endif

// pack two f32 into packed bf16x2 — let the compiler select the hw convert
static __device__ inline unsigned pack2(float a, float b) {
#if HAVE_PK_BF16
  union { v2bf v; unsigned u; } o;
  o.v = __builtin_amdgcn_cvt_pk_bf16_f32(a, b);
  return o.u;
#else
  union { v2bf v; unsigned u; } o;
  o.v = (v2bf){(bf16)a, (bf16)b};      // native f32->bf16 conversions
  return o.u;
#endif
}
static __device__ inline bf16 f2bf(float f) { return (bf16)f; }

// A-matrix (16x32 bf16) per-lane K offset of u32 pair v (ISA 7.12.2):
// v<4 : K = 2v + 8*half ; v>=4 : K = 16 + 2(v-4) + 8*half
static __device__ inline int a_koff(int v, int half) {
  return (v < 4 ? 2 * v : 16 + 2 * (v - 4)) + 8 * half;
}

union Frag {
  v16bf v;
  uint4 q[2];
  unsigned u[8];
};

static __device__ inline v8f wmma_bf16(v16bf a, v16bf b, v8f c) {
  return __builtin_amdgcn_wmma_f32_16x16x32_bf16(false, a, false, b,
                                                 (short)0, c, false, false);
}

// ---------------------------------------------------------------------------
// Kernel 1: QKV projection GEMM (+ fused RMSNorm + RoPE epilogue).
// Block = 256 threads (8 waves, 2x4), tile M=32, N=128 (= one head), K-step 64.
// mode 0/1: norm+rope, row-major [h][s][dh] store.  mode 2: V transposed store.
// ---------------------------------------------------------------------------
__global__ __launch_bounds__(256) void qkv_gemm(
    const float* __restrict__ X, const float* __restrict__ W,
    const float* __restrict__ g, const float* __restrict__ fr,
    bf16* __restrict__ outRM, bf16* __restrict__ outVt,
    int mode, int seqoff)
{
  __shared__ bf16  lds_a[32 * AS];
  __shared__ bf16  lds_bt[128 * BTS];
  __shared__ float lds_c[32 * CS];

  const int tid  = threadIdx.x;
  const int lane = tid & 31;
  const int wave = tid >> 5;
  const int mi   = wave >> 2;     // 0..1
  const int ni   = wave & 3;      // 0..3
  const int half = lane >> 4;
  const int l16  = lane & 15;

  const int m0 = blockIdx.x * 32;
  const int h  = blockIdx.y;          // head index == N tile of 128
  const int nb = h * DH;

  v8f acc0 = {0, 0, 0, 0, 0, 0, 0, 0};
  v8f acc1 = {0, 0, 0, 0, 0, 0, 0, 0};

  unsigned* lau = (unsigned*)lds_a;
  unsigned* lbu = (unsigned*)lds_bt;

  for (int kb = 0; kb < D_MODEL; kb += KSTEP) {
    // --- A tile: X[m0+row][kb..kb+63]  f32 -> bf16 pairs -> LDS ------------
    {
      int row = tid >> 3;
      int c0  = (tid & 7) * 8;
      const float* src = X + (size_t)(m0 + row) * D_MODEL + kb + c0;
      unsigned* dst = lau + row * (AS / 2) + (c0 >> 1);
      dst[0] = pack2(src[0], src[1]);
      dst[1] = pack2(src[2], src[3]);
      dst[2] = pack2(src[4], src[5]);
      dst[3] = pack2(src[6], src[7]);
    }
    // --- B tile: W[kb+2pr(+1)][nb+c] -> bf16 pair (k,k+1) -> LDS^T ---------
    {
      int pr = tid >> 3;              // k pair row: k = 2*pr, 2*pr+1
      int c0 = (tid & 7) * 16;
      const float* s0 = W + (size_t)(kb + 2 * pr) * D_MODEL + nb + c0;
      const float* s1 = s0 + D_MODEL;
      if (kb + KSTEP < D_MODEL)       // pull next weight tile (global_prefetch_b8)
        __builtin_prefetch(s0 + (size_t)KSTEP * D_MODEL, 0, 1);
#pragma unroll
      for (int j = 0; j < 16; j++)
        lbu[(c0 + j) * (BTS / 2) + pr] = pack2(s0[j], s1[j]);
    }
    __syncthreads();

    // --- fragments + WMMA: 2 k-substeps x 2 n-tiles per wave ---------------
#pragma unroll
    for (int s = 0; s < 2; s++) {
      Frag fa;
      int m = mi * 16 + l16;
#pragma unroll
      for (int v = 0; v < 8; v++)
        fa.u[v] = lau[(m * AS + s * 32 + half * 16 + 2 * v) >> 1];
#pragma unroll
      for (int t = 0; t < 2; t++) {
        int n = (ni * 2 + t) * 16 + l16;
        Frag fb;
        int base = (n * BTS + s * 32 + half * 16) >> 1;
#pragma unroll
        for (int j = 0; j < 8; j++) fb.u[j] = lbu[base + j];
        if (t == 0) acc0 = wmma_bf16(fa.v, fb.v, acc0);
        else        acc1 = wmma_bf16(fa.v, fb.v, acc1);
      }
    }
    __syncthreads();
  }

  // --- stage C tile to LDS (C layout: M = r + 8*half, N = l16) -------------
#pragma unroll
  for (int t = 0; t < 2; t++) {
    v8f a = t ? acc1 : acc0;
#pragma unroll
    for (int r = 0; r < 8; r++) {
      int row = mi * 16 + r + 8 * half;
      int col = (ni * 2 + t) * 16 + l16;
      lds_c[row * CS + col] = a[r];
    }
  }
  __syncthreads();

  // --- epilogue: 8 threads per row, 16 cols each ---------------------------
  int row = tid >> 3;
  int oct = tid & 7;
  int sglob = seqoff + m0 + row;
  float* crow = &lds_c[row * CS + oct * 16];

  if (mode == 2) {                    // V: transposed store [h][dh][s]
#pragma unroll
    for (int j = 0; j < 16; j++) {
      int col = oct * 16 + j;
      outVt[((size_t)(h * DH + col)) * S_ALL + sglob] = f2bf(crow[j]);
    }
  } else {                            // Q/K: RMSNorm + RoPE
    float ss = 0.f;
#pragma unroll
    for (int j = 0; j < 16; j++) ss += crow[j] * crow[j];
    ss += __shfl_xor(ss, 1, 8);
    ss += __shfl_xor(ss, 2, 8);
    ss += __shfl_xor(ss, 4, 8);
    float rsig = rsqrtf(ss * (1.f / 128.f) + 1e-5f);
    // freqs_cis: [s][p][j][i] strides 256/4/2/1 ; P = oct*8 + p
    const float* f = fr + (size_t)sglob * 256 + oct * 32;
    unsigned* outp =
        (unsigned*)(outRM + ((size_t)h * S_ALL + sglob) * DH + oct * 16);
#pragma unroll
    for (int p = 0; p < 8; p++) {
      int c0 = oct * 16 + 2 * p;
      float x0 = crow[2 * p]     * rsig * g[c0];
      float x1 = crow[2 * p + 1] * rsig * g[c0 + 1];
      float y0 = f[p * 4 + 0] * x0 + f[p * 4 + 1] * x1;
      float y1 = f[p * 4 + 2] * x0 + f[p * 4 + 3] * x1;
      outp[p] = pack2(y0, y1);        // single global_store_b32
    }
  }
}

// ---------------------------------------------------------------------------
// Kernel 2: flash attention. One wave per (head, 16-query tile); 8 waves/block.
// Score WMMAs: A = Q frag, B = contiguous K rows. PV WMMAs: A = P (via LDS
// re-layout, f32->bf16), B = contiguous V^T rows.
// ---------------------------------------------------------------------------
__global__ __launch_bounds__(256) void flash_attn(
    const bf16* __restrict__ Q, const bf16* __restrict__ Kb,
    const bf16* __restrict__ Vt, bf16* __restrict__ ATT)
{
  __shared__ float lds_p[8][16 * PS];

  const int tid  = threadIdx.x;
  const int lane = tid & 31;
  const int wave = tid >> 5;
  const int half = lane >> 4;
  const int l16  = lane & 15;

  const int w  = blockIdx.x * 8 + wave;
  const int h  = w / (S_ALL / 16);
  const int qt = w % (S_ALL / 16);
  const float scale = 0.08838834764831845f;  // 1/sqrt(128)

  // Q fragments for the 4 K-blocks of Dh=128
  Frag qa[4];
  {
    const unsigned* qp =
        (const unsigned*)(Q + ((size_t)h * S_ALL + qt * 16 + l16) * DH);
#pragma unroll
    for (int kb = 0; kb < 4; kb++)
#pragma unroll
      for (int v = 0; v < 8; v++)
        qa[kb].u[v] = qp[(kb * 32 + half * 16 + 2 * v) >> 1];
  }

  float mrow[8], lrow[8];
  v8f o[8];
#pragma unroll
  for (int r = 0; r < 8; r++) { mrow[r] = -1e30f; lrow[r] = 0.f; }
#pragma unroll
  for (int t = 0; t < 8; t++) o[t] = (v8f){0, 0, 0, 0, 0, 0, 0, 0};

  float* pl = lds_p[wave];

  for (int kc = 0; kc < S_ALL; kc += 32) {
    v8f s0 = {0, 0, 0, 0, 0, 0, 0, 0};
    v8f s1 = {0, 0, 0, 0, 0, 0, 0, 0};
#pragma unroll
    for (int kb = 0; kb < 4; kb++) {
      Frag f0, f1;
      const uint4* k0 = (const uint4*)(Kb + ((size_t)h * S_ALL + kc + l16) * DH +
                                       kb * 32 + half * 16);
      const uint4* k1 = (const uint4*)(Kb + ((size_t)h * S_ALL + kc + 16 + l16) * DH +
                                       kb * 32 + half * 16);
      f0.q[0] = k0[0]; f0.q[1] = k0[1];
      f1.q[0] = k1[0]; f1.q[1] = k1[1];
      s0 = wmma_bf16(qa[kb].v, f0.v, s0);
      s1 = wmma_bf16(qa[kb].v, f1.v, s1);
    }

    // online softmax update (row stats reduced over the 16 lanes of each half)
#pragma unroll
    for (int r = 0; r < 8; r++) {
      float a0 = s0[r] * scale, a1 = s1[r] * scale;
      float cm = fmaxf(a0, a1);
      cm = fmaxf(cm, __shfl_xor(cm, 1, 16));
      cm = fmaxf(cm, __shfl_xor(cm, 2, 16));
      cm = fmaxf(cm, __shfl_xor(cm, 4, 16));
      cm = fmaxf(cm, __shfl_xor(cm, 8, 16));
      float nm    = fmaxf(mrow[r], cm);
      float alpha = __expf(mrow[r] - nm);
      mrow[r] = nm;
      float p0 = __expf(a0 - nm);
      float p1 = __expf(a1 - nm);
      float rs = p0 + p1;
      rs += __shfl_xor(rs, 1, 16);
      rs += __shfl_xor(rs, 2, 16);
      rs += __shfl_xor(rs, 4, 16);
      rs += __shfl_xor(rs, 8, 16);
      lrow[r] = lrow[r] * alpha + rs;
#pragma unroll
      for (int t = 0; t < 8; t++) o[t][r] *= alpha;
      pl[(r + 8 * half) * PS + l16]      = p0;
      pl[(r + 8 * half) * PS + 16 + l16] = p1;
    }
    asm volatile("s_wait_dscnt 0" ::: "memory");

    // re-layout P (C layout -> A layout), f32 -> bf16
    Frag pa;
#pragma unroll
    for (int v = 0; v < 8; v++) {
      int k = half * 16 + 2 * v;
      pa.u[v] = pack2(pl[l16 * PS + k], pl[l16 * PS + k + 1]);
    }

    // PV: B frags are contiguous V^T rows [h][dh][s]
#pragma unroll
    for (int t = 0; t < 8; t++) {
      Frag vb;
      const uint4* vp = (const uint4*)(Vt + ((size_t)h * DH + t * 16 + l16) * S_ALL +
                                       kc + half * 16);
      vb.q[0] = vp[0]; vb.q[1] = vp[1];
      o[t] = wmma_bf16(pa.v, vb.v, o[t]);
    }
  }

  // epilogue: out / l, store bf16 attn[s][h*128+dh]
#pragma unroll
  for (int r = 0; r < 8; r++) lrow[r] = 1.f / lrow[r];
#pragma unroll
  for (int t = 0; t < 8; t++)
#pragma unroll
    for (int r = 0; r < 8; r++) {
      int srow = qt * 16 + r + 8 * half;
      int col  = h * DH + t * 16 + l16;
      ATT[(size_t)srow * D_MODEL + col] = f2bf(o[t][r] * lrow[r]);
    }
}

// ---------------------------------------------------------------------------
// Kernel 3: output projection GEMM (+ bias), f32 output.
// ---------------------------------------------------------------------------
__global__ __launch_bounds__(256) void out_proj(
    const bf16* __restrict__ Att, const float* __restrict__ W,
    const float* __restrict__ bias, float* __restrict__ out, int rowBase)
{
  __shared__ bf16  lds_a[32 * AS];
  __shared__ bf16  lds_bt[128 * BTS];
  __shared__ float lds_c[32 * CS];

  const int tid  = threadIdx.x;
  const int lane = tid & 31;
  const int wave = tid >> 5;
  const int mi   = wave >> 2;
  const int ni   = wave & 3;
  const int half = lane >> 4;
  const int l16  = lane & 15;

  const int m0 = blockIdx.x * 32;
  const int nb = blockIdx.y * 128;

  v8f acc0 = {0, 0, 0, 0, 0, 0, 0, 0};
  v8f acc1 = {0, 0, 0, 0, 0, 0, 0, 0};

  unsigned* lau = (unsigned*)lds_a;
  unsigned* lbu = (unsigned*)lds_bt;

  for (int kb = 0; kb < D_MODEL; kb += KSTEP) {
    {   // A tile (already bf16): 16B per thread
      int row = tid >> 3;
      int c0  = (tid & 7) * 8;
      const uint4* src =
          (const uint4*)(Att + (size_t)(rowBase + m0 + row) * D_MODEL + kb + c0);
      uint4 d = *src;
      unsigned* dst = lau + row * (AS / 2) + (c0 >> 1);
      dst[0] = d.x; dst[1] = d.y; dst[2] = d.z; dst[3] = d.w;
    }
    {   // B tile f32 -> bf16 pairs, transposed
      int pr = tid >> 3;
      int c0 = (tid & 7) * 16;
      const float* s0 = W + (size_t)(kb + 2 * pr) * D_MODEL + nb + c0;
      const float* s1 = s0 + D_MODEL;
      if (kb + KSTEP < D_MODEL)
        __builtin_prefetch(s0 + (size_t)KSTEP * D_MODEL, 0, 1);
#pragma unroll
      for (int j = 0; j < 16; j++)
        lbu[(c0 + j) * (BTS / 2) + pr] = pack2(s0[j], s1[j]);
    }
    __syncthreads();

#pragma unroll
    for (int s = 0; s < 2; s++) {
      Frag fa;
      int m = mi * 16 + l16;
#pragma unroll
      for (int v = 0; v < 8; v++)
        fa.u[v] = lau[(m * AS + s * 32 + half * 16 + 2 * v) >> 1];
#pragma unroll
      for (int t = 0; t < 2; t++) {
        int n = (ni * 2 + t) * 16 + l16;
        Frag fb;
        int base = (n * BTS + s * 32 + half * 16) >> 1;
#pragma unroll
        for (int j = 0; j < 8; j++) fb.u[j] = lbu[base + j];
        if (t == 0) acc0 = wmma_bf16(fa.v, fb.v, acc0);
        else        acc1 = wmma_bf16(fa.v, fb.v, acc1);
      }
    }
    __syncthreads();
  }

#pragma unroll
  for (int t = 0; t < 2; t++) {
    v8f a = t ? acc1 : acc0;
#pragma unroll
    for (int r = 0; r < 8; r++) {
      int row = mi * 16 + r + 8 * half;
      int col = (ni * 2 + t) * 16 + l16;
      lds_c[row * CS + col] = a[r];
    }
  }
  __syncthreads();

  int row = tid >> 3;
  int oct = tid & 7;
  float* crow = &lds_c[row * CS + oct * 16];
#pragma unroll
  for (int j = 0; j < 16; j++) {
    int col = nb + oct * 16 + j;
    out[(size_t)(m0 + row) * D_MODEL + col] = crow[j] + bias[col];
  }
}

// ---------------------------------------------------------------------------
extern "C" void kernel_launch(void* const* d_in, const int* in_sizes, int n_in,
                              void* d_out, int out_size, void* d_ws, size_t ws_size,
                              hipStream_t stream) {
  (void)in_sizes; (void)n_in; (void)out_size; (void)ws_size;
  const float* hs  = (const float*)d_in[0];
  const float* ehs = (const float*)d_in[1];
  const float* fr  = (const float*)d_in[2];
  const float* wq  = (const float*)d_in[3];
  const float* wk  = (const float*)d_in[4];
  const float* wv  = (const float*)d_in[5];
  const float* waq = (const float*)d_in[6];
  const float* wak = (const float*)d_in[7];
  const float* wav = (const float*)d_in[8];
  const float* gq  = (const float*)d_in[9];
  const float* gk  = (const float*)d_in[10];
  const float* gaq = (const float*)d_in[11];
  const float* gak_= (const float*)d_in[12];
  const float* wo  = (const float*)d_in[13];
  const float* bo  = (const float*)d_in[14];
  const float* wao = (const float*)d_in[15];
  const float* bao = (const float*)d_in[16];

  const size_t qsz = (size_t)NH * S_ALL * DH;   // 7.86M bf16 elements
  bf16* Q   = (bf16*)d_ws;
  bf16* Kb  = Q + qsz;
  bf16* Vt  = Kb + qsz;
  bf16* ATT = Vt + qsz;

  dim3 blk(256);
  // img stream occupies rows [512, 2560), txt rows [0, 512) of concat seq
  qkv_gemm<<<dim3(S_IMG / 32, NH), blk, 0, stream>>>(hs,  wq,  gq,  fr, Q,  nullptr, 0, S_TXT);
  qkv_gemm<<<dim3(S_IMG / 32, NH), blk, 0, stream>>>(hs,  wk,  gk,  fr, Kb, nullptr, 1, S_TXT);
  qkv_gemm<<<dim3(S_IMG / 32, NH), blk, 0, stream>>>(hs,  wv,  nullptr, fr, nullptr, Vt, 2, S_TXT);
  qkv_gemm<<<dim3(S_TXT / 32, NH), blk, 0, stream>>>(ehs, waq, gaq, fr, Q,  nullptr, 0, 0);
  qkv_gemm<<<dim3(S_TXT / 32, NH), blk, 0, stream>>>(ehs, wak, gak_, fr, Kb, nullptr, 1, 0);
  qkv_gemm<<<dim3(S_TXT / 32, NH), blk, 0, stream>>>(ehs, wav, nullptr, fr, nullptr, Vt, 2, 0);

  flash_attn<<<dim3(NH * (S_ALL / 16) / 8), blk, 0, stream>>>(Q, Kb, Vt, ATT);

  float* out_img = (float*)d_out;
  float* out_txt = out_img + (size_t)S_IMG * D_MODEL;
  out_proj<<<dim3(S_IMG / 32, NH), blk, 0, stream>>>(ATT, wo,  bo,  out_img, S_TXT);
  out_proj<<<dim3(S_TXT / 32, NH), blk, 0, stream>>>(ATT, wao, bao, out_txt, 0);
}
